// RetrosynthesisSeq2SeqModel_90701119356943
// MI455X (gfx1250) — matrix operand order repaired and checked by reading, not compiled, not run on hardware
//
#include <hip/hip_runtime.h>

typedef __attribute__((ext_vector_type(16))) _Float16 v16h;
typedef __attribute__((ext_vector_type(8)))  _Float16 v8h;
typedef __attribute__((ext_vector_type(8)))  float    v8f;

__device__ __forceinline__ float sigmoidf_(float x) { return 1.0f / (1.0f + __expf(-x)); }

// ---------------------------------------------------------------------------
// Weight swizzle: f32 row-major [K][N] -> f16 WMMA B-fragment order.
// Fragment (tk, tn) holds K rows [tk*32, tk*32+32), cols [tn*16, tn*16+16):
//   dst[(((tk*(N/16)) + tn)*32 + lane)*16 + j] = W[tk*32 + (j&8)*2 + (lane>>4)*8 + (j&7)][tn*16 + (lane&15)]
// so each lane's 16 halfwords (32 B) are contiguous -> single wide load.
// ---------------------------------------------------------------------------
__global__ void k_swz_weight(const float* __restrict__ W, _Float16* __restrict__ dst,
                             int K, int N) {
    int i = blockIdx.x * blockDim.x + threadIdx.x;
    if (i >= K * N) return;
    int j    = i & 15;
    int lane = (i >> 4) & 31;
    int tile = i >> 9;
    int tilesN = N >> 4;
    int tn = tile % tilesN;
    int tk = tile / tilesN;
    int srck = tk * 32 + (j & 8) * 2 + (lane >> 4) * 8 + (j & 7);
    int srcn = tn * 16 + (lane & 15);
    dst[i] = (_Float16)W[(size_t)srck * N + srcn];
}

// ---------------------------------------------------------------------------
// Embedding gather, TIME-MAJOR output: out[(t*16+b)*edim + j] = emb[ids[b*T+t]][j]
// ---------------------------------------------------------------------------
__global__ void k_embed_tm(const int* __restrict__ ids, const float* __restrict__ emb,
                           float* __restrict__ out, int T, int edim, int ntotal) {
    int i = blockIdx.x * blockDim.x + threadIdx.x;
    if (i >= ntotal) return;
    int row = i / edim, j = i - row * edim;
    int b = row & 15, t = row >> 4;
    out[i] = emb[ids[b * T + t] * edim + j];
}

// ---------------------------------------------------------------------------
// WMMA GEMM: C[M,N] = A[M,K](f32) @ Bsw[K,N](f16, pre-swizzled fragments) + bias
// One 16x16 tile per wave32. If swz_out: C stored in accumulator-fragment
// order (lane-contiguous v8f); else row-major f32.
// ---------------------------------------------------------------------------
__global__ __launch_bounds__(256) void k_gemm(const float* __restrict__ A,
                                              const _Float16* __restrict__ Bsw,
                                              const float* __restrict__ bias,
                                              float* __restrict__ C,
                                              int M, int N, int K, int lda, int ldc,
                                              int swz_out) {
    int gw = (blockIdx.x * blockDim.x + threadIdx.x) >> 5;   // wave-uniform
    int tilesN = N >> 4;
    int tileM = gw / tilesN;
    int tileN = gw - tileM * tilesN;
    if (tileM * 16 >= M) return;                              // wave-uniform exit
    int lane = threadIdx.x & 31;
    int half = lane >> 4;
    int l16  = lane & 15;

    const float* arow = A + (size_t)(tileM * 16 + l16) * lda;
    int coln = tileN * 16 + l16;
    float bb = bias ? bias[coln] : 0.0f;

    v8f acc = {};
    for (int k0 = 0; k0 < K; k0 += 32) {
        // A 16x32 f16 fragment: lane<16 K=[k0..k0+7]+[k0+16..k0+23]; lane>=16 +8
        v8f alo = *(const v8f*)(arow + k0 + half * 8);
        v8f ahi = *(const v8f*)(arow + k0 + 16 + half * 8);
        v16h a;
#pragma unroll
        for (int j = 0; j < 8; ++j) {
            a[j]     = (_Float16)alo[j];
            a[j + 8] = (_Float16)ahi[j];
        }
        // B fragment: contiguous 32 B per lane from swizzled weights
        const _Float16* bp = Bsw + ((((size_t)(k0 >> 5)) * tilesN + tileN) * 32 + lane) * 16;
        __builtin_prefetch(bp + (size_t)tilesN * 512, 0, 1);  // next k-panel
        v16h b = *(const v16h*)bp;
        acc = __builtin_amdgcn_wmma_f32_16x16x32_f16(false, a, false, b, (short)0, acc,
                                                     false, false);
    }
#pragma unroll
    for (int r = 0; r < 8; ++r) acc[r] += bb;
    if (swz_out) {
        float* cp = C + ((((size_t)tileM) * tilesN + tileN) * 32 + lane) * 8;
        *(v8f*)cp = acc;
    } else {
#pragma unroll
        for (int r = 0; r < 8; ++r) {
            int row = tileM * 16 + r + half * 8;
            C[(size_t)row * ldc + coln] = acc[r];
        }
    }
}

// ---------------------------------------------------------------------------
// Persistent single-workgroup masked LSTM scan (Keras semantics), wave32 x16.
// Wave w owns hidden units [32w, 32w+32) across all 4 gates (8 WMMA tiles).
// h: LDS f16 (WMMA A operand). c: accumulator-layout VGPRs.
// XWf: x@W+b in accumulator-fragment order, tile (t, n-tile of 2048).
// Uwsw: f16 swizzled [512][2048]. Y written TIME-MAJOR [(t*16+b)][ldy].
// ---------------------------------------------------------------------------
__global__ __launch_bounds__(512) void k_lstm(const float* __restrict__ XWf,
                                              const _Float16* __restrict__ Uwsw,
                                              const int* __restrict__ seq,  // [16][T]
                                              const float* __restrict__ h0, // [16][512]|null
                                              const float* __restrict__ c0, // [16][512]|null
                                              float* __restrict__ Y, int ldy, int ycol0,
                                              float* __restrict__ hT, float* __restrict__ cT,
                                              int hstride, int T, int reverse) {
    __shared__ __align__(32) _Float16 hS[16][512];
    const int tid  = threadIdx.x;
    const int wave = tid >> 5;
    const int lane = tid & 31;
    const int half = lane >> 4;
    const int l16  = lane & 15;
    const int u0   = wave * 32;

    for (int i = tid; i < 16 * 512; i += 512) {
        int r = i >> 9, c = i & 511;
        hS[r][c] = h0 ? (_Float16)h0[r * 512 + c] : (_Float16)0.0f;
    }
    v8f creg[2];
#pragma unroll
    for (int ut = 0; ut < 2; ++ut) {
        int col = u0 + ut * 16 + l16;
#pragma unroll
        for (int r = 0; r < 8; ++r)
            creg[ut][r] = c0 ? c0[(r + half * 8) * 512 + col] : 0.0f;
    }
    __syncthreads();

    for (int t = 0; t < T; ++t) {
        const int tt = reverse ? (T - 1 - t) : t;
        v8f acc[4][2];
        // gate pre-activations from fragment-layout XW (bias already fused)
#pragma unroll
        for (int g = 0; g < 4; ++g)
#pragma unroll
            for (int ut = 0; ut < 2; ++ut) {
                int tIdx = g * 32 + wave * 2 + ut;                 // n-tile of 2048
                const float* fp = XWf + ((((size_t)tt) * 128 + tIdx) * 32 + lane) * 8;
                acc[g][ut] = *(const v8f*)fp;
            }
        // z += h @ Uw : K=512 -> 16 k-steps x 8 v_wmma_f32_16x16x32_f16
        for (int k0 = 0; k0 < 512; k0 += 32) {
            v8h alo = *(const v8h*)(&hS[l16][k0 + half * 8]);
            v8h ahi = *(const v8h*)(&hS[l16][k0 + 16 + half * 8]);
            v16h a = __builtin_shufflevector(alo, ahi, 0, 1, 2, 3, 4, 5, 6, 7,
                                             8, 9, 10, 11, 12, 13, 14, 15);
#pragma unroll
            for (int g = 0; g < 4; ++g)
#pragma unroll
                for (int ut = 0; ut < 2; ++ut) {
                    int tIdx = g * 32 + wave * 2 + ut;
                    const _Float16* bp =
                        Uwsw + ((((size_t)(k0 >> 5)) * 128 + tIdx) * 32 + lane) * 16;
                    v16h b = *(const v16h*)bp;
                    acc[g][ut] = __builtin_amdgcn_wmma_f32_16x16x32_f16(
                        false, a, false, b, (short)0, acc[g][ut], false, false);
                }
        }
        // gates + masked state update
        float hnew[2][8];
#pragma unroll
        for (int ut = 0; ut < 2; ++ut) {
            int col = u0 + ut * 16 + l16;
#pragma unroll
            for (int r = 0; r < 8; ++r) {
                int row  = r + half * 8;                           // batch index
                float ig = sigmoidf_(acc[0][ut][r]);
                float fg = sigmoidf_(acc[1][ut][r]);
                float gg = tanhf(acc[2][ut][r]);
                float og = sigmoidf_(acc[3][ut][r]);
                float cn = fg * creg[ut][r] + ig * gg;
                float hn = og * tanhf(cn);
                bool  m  = seq[row * T + tt] != 0;
                float hp = (float)hS[row][col];
                float h2 = m ? hn : hp;
                creg[ut][r] = m ? cn : creg[ut][r];
                hnew[ut][r] = h2;
                Y[(size_t)(tt * 16 + row) * ldy + ycol0 + col] = h2;  // time-major
            }
        }
        __syncthreads();
#pragma unroll
        for (int ut = 0; ut < 2; ++ut) {
            int col = u0 + ut * 16 + l16;
#pragma unroll
            for (int r = 0; r < 8; ++r) hS[r + half * 8][col] = (_Float16)hnew[ut][r];
        }
        __syncthreads();
    }
    if (hT)
        for (int i = tid; i < 16 * 512; i += 512) {
            int r = i >> 9, c = i & 511;
            hT[r * hstride + c] = (float)hS[r][c];
        }
    if (cT) {
#pragma unroll
        for (int ut = 0; ut < 2; ++ut) {
            int col = u0 + ut * 16 + l16;
#pragma unroll
            for (int r = 0; r < 8; ++r) cT[(r + half * 8) * hstride + col] = creg[ut][r];
        }
    }
}

// ---------------------------------------------------------------------------
// Fused attention (time-major rows: seq index s, batch b -> row s*16+b).
// One block per (b,td): e over te, softmax, context; writes concat row
// [dec_out(512) | context(1024)] at row td*16+b.
// ---------------------------------------------------------------------------
__global__ __launch_bounds__(128) void k_attention(const float* __restrict__ EP,   // [te*16+b][512]
                                                   const float* __restrict__ DP,   // [td*16+b][512]
                                                   const float* __restrict__ encout, // [te*16+b][1024]
                                                   const float* __restrict__ decout, // [td*16+b][512]
                                                   const float* __restrict__ vW,
                                                   const float* __restrict__ vB,
                                                   float* __restrict__ CC) {       // [td*16+b][1536]
    __shared__ float dp[512], vv[512], e[128], red[128];
    int b  = blockIdx.x >> 7;
    int td = blockIdx.x & 127;
    int tid = threadIdx.x;
    int drow = td * 16 + b;

    for (int j = tid; j < 512; j += 128) { dp[j] = DP[(size_t)drow * 512 + j]; vv[j] = vW[j]; }
    __syncthreads();

    const float* ep = EP + (size_t)(tid * 16 + b) * 512;   // te = tid
    float acc = 0.0f;
    for (int u = 0; u < 512; ++u) acc += vv[u] * tanhf(ep[u] + dp[u]);
    e[tid] = acc + vB[0];
    __syncthreads();

    red[tid] = e[tid];
    __syncthreads();
    for (int s = 64; s > 0; s >>= 1) { if (tid < s) red[tid] = fmaxf(red[tid], red[tid + s]); __syncthreads(); }
    float mx = red[0];
    __syncthreads();
    float myexp = __expf(e[tid] - mx);
    red[tid] = myexp;
    __syncthreads();
    for (int s = 64; s > 0; s >>= 1) { if (tid < s) red[tid] += red[tid + s]; __syncthreads(); }
    float sum = red[0];
    __syncthreads();
    e[tid] = myexp / sum;
    __syncthreads();

    float* ccrow = CC + (size_t)drow * 1536;
    for (int c = tid; c < 1024; c += 128) {
        float s = 0.0f;
        const float* eo = encout + (size_t)b * 1024 + c;
        for (int te = 0; te < 128; ++te) s += e[te] * eo[(size_t)te * 16 * 1024];
        ccrow[512 + c] = s;
    }
    for (int c = tid; c < 512; c += 128) ccrow[c] = decout[(size_t)drow * 512 + c];
}

// ---------------------------------------------------------------------------
// Row softmax over 64 logits; input rows time-major (td*16+b), output [b][td][64]
// ---------------------------------------------------------------------------
__global__ void k_softmax64(const float* __restrict__ L, float* __restrict__ O, int rows) {
    int r = blockIdx.x * blockDim.x + threadIdx.x;
    if (r >= rows) return;
    int b = r >> 7, td = r & 127;
    const float* l = L + (size_t)(td * 16 + b) * 64;
    float* o = O + (size_t)r * 64;
    float mx = -1e30f;
    for (int j = 0; j < 64; ++j) mx = fmaxf(mx, l[j]);
    float s = 0.0f;
    for (int j = 0; j < 64; ++j) { float v = __expf(l[j] - mx); o[j] = v; s += v; }
    float inv = 1.0f / s;
    for (int j = 0; j < 64; ++j) o[j] *= inv;
}

// ---------------------------------------------------------------------------
// Host orchestration
// ---------------------------------------------------------------------------
extern "C" void kernel_launch(void* const* d_in, const int* in_sizes, int n_in,
                              void* d_out, int out_size, void* d_ws, size_t ws_size,
                              hipStream_t stream) {
    (void)in_sizes; (void)n_in; (void)out_size; (void)ws_size;
    const int B = 16, TE = 128, TD = 128, EMB = 256, U = 512, NG = 2048, VOUT = 64;
    const int M = B * TE;  // 2048

    const int* enc_ids = (const int*)d_in[0];
    const int* dec_ids = (const int*)d_in[1];
    const float* emb_enc = (const float*)d_in[2];
    const float* emb_dec = (const float*)d_in[3];
    const float* lW[8]; const float* lU[8]; const float* lb[8]; int lK[8];
    for (int i = 0; i < 8; ++i) {
        lW[i] = (const float*)d_in[4 + 3 * i];
        lU[i] = (const float*)d_in[5 + 3 * i];
        lb[i] = (const float*)d_in[6 + 3 * i];
    }
    lK[0] = EMB; lK[1] = EMB;
    lK[2] = 2 * U; lK[3] = 2 * U;
    lK[4] = EMB; lK[5] = U; lK[6] = U; lK[7] = U;
    const float* projhW = (const float*)d_in[28]; const float* projhB = (const float*)d_in[29];
    const float* projcW = (const float*)d_in[30]; const float* projcB = (const float*)d_in[31];
    const float* att1W  = (const float*)d_in[32]; const float* att1B  = (const float*)d_in[33];
    const float* att2W  = (const float*)d_in[34]; const float* att2B  = (const float*)d_in[35];
    const float* attvW  = (const float*)d_in[36]; const float* attvB  = (const float*)d_in[37];
    const float* outW   = (const float*)d_in[38]; const float* outB   = (const float*)d_in[39];

    char* ws = (char*)d_ws;
    size_t off = 0;
    auto alloc = [&](size_t bytes) -> void* {
        void* p = ws + off;
        off = (off + bytes + 255) & ~(size_t)255;
        return p;
    };
    auto swz = [&](const float* src, int K, int N) -> _Float16* {
        _Float16* d = (_Float16*)alloc((size_t)K * N * sizeof(_Float16));
        k_swz_weight<<<((K * N) + 255) / 256, 256, 0, stream>>>(src, d, K, N);
        return d;
    };

    // swizzled f16 weights
    _Float16* lW16[8]; _Float16* lU16[8];
    for (int i = 0; i < 8; ++i) {
        lW16[i] = swz(lW[i], lK[i], NG);
        lU16[i] = swz(lU[i], U, NG);
    }
    _Float16* projh16 = swz(projhW, 2 * U, U);
    _Float16* projc16 = swz(projcW, 2 * U, U);
    _Float16* att1_16 = swz(att1W, 2 * U, U);
    _Float16* att2_16 = swz(att2W, U, U);
    _Float16* out16   = swz(outW, 3 * U, VOUT);

    // f32 activation buffers (sequence buffers are TIME-MAJOR)
    float* X0e  = (float*)alloc((size_t)M * EMB * 4);
    float* X0d  = (float*)alloc((size_t)M * EMB * 4);
    float* XW   = (float*)alloc((size_t)M * NG * 4);   // accumulator-fragment layout
    float* Y1   = (float*)alloc((size_t)M * 2 * U * 4);
    float* Y2   = (float*)alloc((size_t)M * 2 * U * 4);
    float* D1   = (float*)alloc((size_t)M * U * 4);
    float* D2   = (float*)alloc((size_t)M * U * 4);
    float* hcat = (float*)alloc((size_t)B * 2 * U * 4);
    float* ccat = (float*)alloc((size_t)B * 2 * U * 4);
    float* dh0  = (float*)alloc((size_t)B * U * 4);
    float* dc0  = (float*)alloc((size_t)B * U * 4);
    float* EP   = (float*)alloc((size_t)M * U * 4);
    float* DP   = (float*)alloc((size_t)M * U * 4);
    float* CC   = (float*)alloc((size_t)M * 3 * U * 4);
    float* LG   = (float*)alloc((size_t)M * VOUT * 4);

    auto gemm = [&](const float* A, const _Float16* Bw, const float* bias, float* C,
                    int m, int n, int k, int lda, int ldc, int swz_out) {
        int tiles = (m / 16) * (n / 16);
        int blocks = (tiles * 32 + 255) / 256;
        k_gemm<<<blocks, 256, 0, stream>>>(A, Bw, bias, C, m, n, k, lda, ldc, swz_out);
    };

    // embeddings (time-major)
    k_embed_tm<<<(M * EMB + 255) / 256, 256, 0, stream>>>(enc_ids, emb_enc, X0e, TE, EMB, M * EMB);
    k_embed_tm<<<(M * EMB + 255) / 256, 256, 0, stream>>>(dec_ids, emb_dec, X0d, TD, EMB, M * EMB);

    // encoder layer 0 (bidirectional)
    gemm(X0e, lW16[0], lb[0], XW, M, NG, EMB, EMB, NG, 1);
    k_lstm<<<1, 512, 0, stream>>>(XW, lU16[0], enc_ids, nullptr, nullptr,
                                  Y1, 2 * U, 0, nullptr, nullptr, 0, TE, 0);
    gemm(X0e, lW16[1], lb[1], XW, M, NG, EMB, EMB, NG, 1);
    k_lstm<<<1, 512, 0, stream>>>(XW, lU16[1], enc_ids, nullptr, nullptr,
                                  Y1, 2 * U, U, nullptr, nullptr, 0, TE, 1);

    // encoder layer 1 (bidirectional, captures final states)
    gemm(Y1, lW16[2], lb[2], XW, M, NG, 2 * U, 2 * U, NG, 1);
    k_lstm<<<1, 512, 0, stream>>>(XW, lU16[2], enc_ids, nullptr, nullptr,
                                  Y2, 2 * U, 0, hcat, ccat, 2 * U, TE, 0);
    gemm(Y1, lW16[3], lb[3], XW, M, NG, 2 * U, 2 * U, NG, 1);
    k_lstm<<<1, 512, 0, stream>>>(XW, lU16[3], enc_ids, nullptr, nullptr,
                                  Y2, 2 * U, U, hcat + U, ccat + U, 2 * U, TE, 1);

    // state projections (row-major out)
    gemm(hcat, projh16, projhB, dh0, B, U, 2 * U, 2 * U, U, 0);
    gemm(ccat, projc16, projcB, dc0, B, U, 2 * U, 2 * U, U, 0);

    // decoder stack
    gemm(X0d, lW16[4], lb[4], XW, M, NG, EMB, EMB, NG, 1);
    k_lstm<<<1, 512, 0, stream>>>(XW, lU16[4], dec_ids, dh0, dc0,
                                  D1, U, 0, nullptr, nullptr, 0, TD, 0);
    gemm(D1, lW16[5], lb[5], XW, M, NG, U, U, NG, 1);
    k_lstm<<<1, 512, 0, stream>>>(XW, lU16[5], dec_ids, nullptr, nullptr,
                                  D2, U, 0, nullptr, nullptr, 0, TD, 0);
    gemm(D2, lW16[6], lb[6], XW, M, NG, U, U, NG, 1);
    k_lstm<<<1, 512, 0, stream>>>(XW, lU16[6], dec_ids, nullptr, nullptr,
                                  D1, U, 0, nullptr, nullptr, 0, TD, 0);
    gemm(D1, lW16[7], lb[7], XW, M, NG, U, U, NG, 1);
    k_lstm<<<1, 512, 0, stream>>>(XW, lU16[7], dec_ids, nullptr, nullptr,
                                  D2, U, 0, nullptr, nullptr, 0, TD, 0);

    // attention
    gemm(Y2, att1_16, att1B, EP, M, U, 2 * U, 2 * U, U, 0);
    gemm(D2, att2_16, att2B, DP, M, U, U, U, U, 0);
    k_attention<<<B * TD, 128, 0, stream>>>(EP, DP, Y2, D2, attvW, attvB, CC);

    // output projection + softmax
    gemm(CC, out16, outB, LG, M, VOUT, 3 * U, 3 * U, VOUT, 0);
    k_softmax64<<<(M + 127) / 128, 128, 0, stream>>>(LG, (float*)d_out, M);
}
